// ChannelSpatialModulatedConv2d_76948634075561
// MI455X (gfx1250) — matrix operand order repaired and checked by reading, not compiled, not run on hardware
//
#include <hip/hip_runtime.h>
#include <hip/hip_bf16.h>
#include <math.h>

// ---------------------------------------------------------------------------
// Problem constants (match reference)
// ---------------------------------------------------------------------------
#define BATCH   32
#define CIN     256
#define COUT    256
#define KSZ     3
#define STYLE   512
#define SPD     64           // spatial dim
#define NPIX    (SPD*SPD)    // 4096
#define EPS     1e-6f

typedef __bf16 bf16_t;
typedef __attribute__((ext_vector_type(16))) __bf16 v16bf;
typedef __attribute__((ext_vector_type(8)))  __bf16 v8bf;
typedef __attribute__((ext_vector_type(4)))  __bf16 v4bf;
typedef __attribute__((ext_vector_type(8)))  float  v8f;

// ---------------------------------------------------------------------------
// Kernel 1: channel style = style_chan @ (mod_w * lin_scale)^T + mod_b
// grid = B, block = 256 (one thread per ci)
// ---------------------------------------------------------------------------
__global__ __launch_bounds__(256)
void style_kernel(const float* __restrict__ style_in,
                  const float* __restrict__ mod_w,
                  const float* __restrict__ mod_b,
                  float* __restrict__ style_ws)
{
    __shared__ float sc[CIN];
    const int b  = blockIdx.x;
    const int ci = threadIdx.x;
    sc[ci] = style_in[b * STYLE + ci];          // first half = style_chan
    __syncthreads();

    const float lin_scale = 1.0f / sqrtf((float)(STYLE / 2));   // 1/16
    float acc = 0.0f;
    const float* wrow = mod_w + (size_t)ci * CIN;
    #pragma unroll 8
    for (int j = 0; j < CIN; ++j) acc += sc[j] * wrow[j];
    style_ws[b * CIN + ci] = acc * lin_scale + mod_b[ci];
}

// ---------------------------------------------------------------------------
// Kernel 2: spatial modulation map (incl. demod_sp), stored [B][4096] f32
// grid = B, block = 256 (each thread handles 16 pixels)
// ---------------------------------------------------------------------------
__global__ __launch_bounds__(256)
void smap_kernel(const float* __restrict__ style_in,
                 const float* __restrict__ sp_w,
                 const float* __restrict__ sp_b,
                 float* __restrict__ smap)
{
    __shared__ float ss[CIN];
    __shared__ float red[256];
    const int b = blockIdx.x;
    const int t = threadIdx.x;
    ss[t] = style_in[b * STYLE + (STYLE / 2) + t];   // second half = style_spatial
    __syncthreads();

    const float lin_scale = 1.0f / sqrtf((float)(STYLE / 2));
    float spv[16];
    float ssq = 0.0f;
    #pragma unroll
    for (int it = 0; it < 16; ++it) {
        const int p = t + it * 256;
        const float* wrow = sp_w + (size_t)p * (STYLE / 2);
        float acc = 0.0f;
        #pragma unroll 8
        for (int j = 0; j < STYLE / 2; ++j) acc += ss[j] * wrow[j];
        const float v = acc * lin_scale + sp_b[p];
        spv[it] = v;
        ssq += v * v;
    }
    red[t] = ssq;
    __syncthreads();
    for (int s = 128; s > 0; s >>= 1) {
        if (t < s) red[t] += red[t + s];
        __syncthreads();
    }
    const float scale = sqrtf((float)NPIX / red[0] + EPS);
    #pragma unroll
    for (int it = 0; it < 16; ++it)
        smap[(size_t)b * NPIX + t + it * 256] = spv[it] * scale;
}

// ---------------------------------------------------------------------------
// Kernel 3: modulate + demodulate weights -> bf16, layout [B][9][COUT][CIN]
// grid = (COUT, B), block = 256 (one thread per ci)
// ---------------------------------------------------------------------------
__global__ __launch_bounds__(256)
void wmod_kernel(const float* __restrict__ weight,   // [COUT][CIN][3][3]
                 const float* __restrict__ style_ws, // [B][CIN]
                 bf16_t* __restrict__ wmod)          // [B][9][COUT][CIN]
{
    __shared__ float red[256];
    const int co = blockIdx.x;
    const int b  = blockIdx.y;
    const int ci = threadIdx.x;

    const float conv_scale = 1.0f / sqrtf((float)(CIN * KSZ * KSZ)); // 1/48
    const float s = style_ws[b * CIN + ci];

    float w9[9];
    float ssq = 0.0f;
    const float* wp = weight + ((size_t)co * CIN + ci) * 9;
    #pragma unroll
    for (int j = 0; j < 9; ++j) {
        const float w = conv_scale * wp[j] * s;
        w9[j] = w;
        ssq += w * w;
    }
    red[ci] = ssq;
    __syncthreads();
    for (int sdist = 128; sdist > 0; sdist >>= 1) {
        if (ci < sdist) red[ci] += red[ci + sdist];
        __syncthreads();
    }
    const float demod = rsqrtf(red[0] + EPS);
    #pragma unroll
    for (int j = 0; j < 9; ++j)
        wmod[(((size_t)b * 9 + j) * COUT + co) * CIN + ci] = (bf16_t)(w9[j] * demod);
}

// ---------------------------------------------------------------------------
// Kernel 4: one-time x f32 -> bf16 conversion (halves all conv-side L2 reads)
// ---------------------------------------------------------------------------
__global__ __launch_bounds__(256)
void xcvt_kernel(const float* __restrict__ x, bf16_t* __restrict__ xbf)
{
    const size_t i = ((size_t)blockIdx.x * 256 + threadIdx.x) * 4;
    const float4 v = *(const float4*)(x + i);
    v4bf o = { (bf16_t)v.x, (bf16_t)v.y, (bf16_t)v.z, (bf16_t)v.w };
    *(v4bf*)(xbf + i) = o;
}

// ---------------------------------------------------------------------------
// Kernel 5: grouped conv as per-batch implicit GEMM with bf16 WMMA.
//   WG tile: M=128 (COUT) x N=256 (4 image rows).  K = 9 taps x 256 ci, in
//   72 stages of K=32.  8 waves: 2(M) x 4(N), each wave 64x64 = 4x4 WMMAs.
//   Software pipeline: ds_store(s) -> barrier -> fetch(s+1) -> compute(s),
//   double-buffered LDS (one barrier per stage; reads of buf p at stage s and
//   writes at stage s+2 are separated by the stage s+1 barrier).
//   LDS tiles stored directly in ISA 7.12.2 fragment layouts:
//     A (16x32 bf16): lane = m%16 + 16*((k>>3)&1), elem i = 8*(k>>4)+(k&7)
//     B (32x16 bf16): lane = n%16 + 16*((k>>3)&1), elem i = 8*(k>>4)+(k&7)
//     D (16x16 f32) : lane = n%16 + 16*(m/8),      vgpr v = m%8
// grid = (16 n-tiles, 2 m-tiles, 32 batch), block = 256
// ---------------------------------------------------------------------------
__global__ __launch_bounds__(256)
void conv_wmma_kernel(const bf16_t* __restrict__ xbf,   // [B][CIN][64][64]
                      const bf16_t* __restrict__ wmod,  // [B][9][COUT][CIN]
                      const float* __restrict__ smap,   // [B][4096]
                      float* __restrict__ out)          // [B][COUT][64][64]
{
    __shared__ alignas(32) bf16_t lA[2][8][32][16];    // 2 x  8 KB
    __shared__ alignas(32) bf16_t lB[2][16][32][16];   // 2 x 16 KB

    const int t     = threadIdx.x;
    const int lane  = t & 31;
    const int wid   = t >> 5;
    const int waveM = wid & 1;      // 0..1  -> 64 rows of M each
    const int waveN = wid >> 1;     // 0..3  -> 64 cols of N each

    const int tileN = blockIdx.x;          // pixel tile: rows 4*tileN .. +3
    const int m0    = blockIdx.y * 128;    // COUT tile base
    const int b     = blockIdx.z;
    const int h0    = tileN * 4;

    // staging roles
    const int am   = t >> 1;       // 0..127 : A row (co offset)
    const int akc  = t & 1;        // 0..1   : A 16-wide k chunk
    const int bn   = t;            // 0..255 : B column (pixel in tile)
    const int brow = bn >> 6;      // 0..3
    const int bcol = bn & 63;

    v8f acc[4][4];
    #pragma unroll
    for (int i = 0; i < 4; ++i)
        #pragma unroll
        for (int j = 0; j < 4; ++j)
            acc[i][j] = (v8f){0.f,0.f,0.f,0.f,0.f,0.f,0.f,0.f};

    const size_t xbatch = (size_t)b * CIN * NPIX;

    // register staging for the software pipeline
    v8bf   aLo, aHi;
    bf16_t bv[32];

    auto fetch = [&](int s) {
        const int r  = s >> 3;               // tap 0..8
        const int cb = s & 7;                // ci block 0..7
        const int dh = r / 3 - 1;
        const int dw = r % 3 - 1;

        // ---- A: 16 contiguous bf16 weights (one 32B chunk / thread) ----
        const bf16_t* wsrc =
            wmod + (((size_t)b * 9 + r) * COUT + (m0 + am)) * CIN + cb * 32 + akc * 16;
        aLo = ((const v8bf*)wsrc)[0];        // k = akc*16 + 0..7
        aHi = ((const v8bf*)wsrc)[1];        // k = akc*16 + 8..15
        // prefetch the next co row at the same tap/ci block into WGP cache (streamed once)
        if (s + 1 < 72)
            __builtin_prefetch(wsrc + CIN, 0, 1);

        // ---- B: one pixel / thread, 32 channels (uniform predicate) ----
        const int srow = h0 + brow + dh;
        const int scol = bcol + dw;
        const bool inb = ((unsigned)srow < (unsigned)SPD) &&
                         ((unsigned)scol < (unsigned)SPD);
        if (inb) {
            const bf16_t* xp = xbf + xbatch + (size_t)(cb * 32) * NPIX
                                   + srow * SPD + scol;
            #pragma unroll
            for (int k = 0; k < 32; ++k)
                bv[k] = xp[(size_t)k * NPIX];
        } else {
            #pragma unroll
            for (int k = 0; k < 32; ++k)
                bv[k] = (bf16_t)0.0f;
        }
    };

    auto stage_store = [&](int buf) {
        // A fragment order
        *(v8bf*)&lA[buf][am >> 4][ am & 15      ][8 * akc] = aLo;
        *(v8bf*)&lA[buf][am >> 4][(am & 15) + 16][8 * akc] = aHi;
        // B fragment order: lane = n%16 + 16*((k>>3)&1), i = 8*(k>>4)+(k&7)
        *(v8bf*)&lB[buf][bn >> 4][ bn & 15      ][0] = *(v8bf*)&bv[0];
        *(v8bf*)&lB[buf][bn >> 4][ bn & 15      ][8] = *(v8bf*)&bv[16];
        *(v8bf*)&lB[buf][bn >> 4][(bn & 15) + 16][0] = *(v8bf*)&bv[8];
        *(v8bf*)&lB[buf][bn >> 4][(bn & 15) + 16][8] = *(v8bf*)&bv[24];
    };

    fetch(0);
    for (int s = 0; s < 72; ++s) {
        const int buf = s & 1;
        stage_store(buf);
        __syncthreads();
        if (s < 71) fetch(s + 1);            // overlaps with WMMAs below

        v16bf af[4], bfr[4];
        #pragma unroll
        for (int i = 0; i < 4; ++i)
            af[i] = *(const v16bf*)&lA[buf][waveM * 4 + i][lane][0];
        #pragma unroll
        for (int j = 0; j < 4; ++j)
            bfr[j] = *(const v16bf*)&lB[buf][waveN * 4 + j][lane][0];

        #pragma unroll
        for (int i = 0; i < 4; ++i)
            #pragma unroll
            for (int j = 0; j < 4; ++j)
                acc[i][j] = __builtin_amdgcn_wmma_f32_16x16x32_bf16(
                    false, af[i], false, bfr[j],
                    (short)0, acc[i][j], false, false);
    }

    // ---- epilogue: spatial modulation + store ----
    #pragma unroll
    for (int j = 0; j < 4; ++j) {
        const int p = tileN * 256 + waveN * 64 + j * 16 + (lane & 15);
        const float smv = smap[(size_t)b * NPIX + p];
        #pragma unroll
        for (int i = 0; i < 4; ++i) {
            const int mbase = m0 + waveM * 64 + i * 16 + 8 * (lane >> 4);
            #pragma unroll
            for (int v = 0; v < 8; ++v) {
                const int m = mbase + v;
                out[((size_t)b * COUT + m) * NPIX + p] = acc[i][j][v] * smv;
            }
        }
    }
}

// ---------------------------------------------------------------------------
// Launch
// ---------------------------------------------------------------------------
extern "C" void kernel_launch(void* const* d_in, const int* in_sizes, int n_in,
                              void* d_out, int out_size, void* d_ws, size_t ws_size,
                              hipStream_t stream)
{
    (void)in_sizes; (void)n_in; (void)out_size; (void)ws_size;

    const float* x        = (const float*)d_in[0];   // [32,256,64,64]
    const float* style_in = (const float*)d_in[1];   // [32,512]
    const float* weight   = (const float*)d_in[2];   // [1,256,256,3,3]
    const float* mod_w    = (const float*)d_in[3];   // [256,256]
    const float* mod_b    = (const float*)d_in[4];   // [256]
    const float* sp_w     = (const float*)d_in[5];   // [4096,256]
    const float* sp_b     = (const float*)d_in[6];   // [4096]
    float* out            = (float*)d_out;           // [32,256,64,64]

    // workspace layout
    char*   ws       = (char*)d_ws;
    float*  style_ws = (float*)ws;                           //  32 KB  [B][CIN]
    float*  smap     = (float*)(ws + 32768);                 // 512 KB  [B][4096]
    bf16_t* wmod     = (bf16_t*)(ws + 32768 + 524288);       // 37.75 MB [B][9][CO][CI]
    bf16_t* xbf      = (bf16_t*)(ws + 32768 + 524288 + 37748736); // 67 MB [B][CIN][64][64]

    style_kernel<<<BATCH, 256, 0, stream>>>(style_in, mod_w, mod_b, style_ws);
    smap_kernel<<<BATCH, 256, 0, stream>>>(style_in, sp_w, sp_b, smap);
    wmod_kernel<<<dim3(COUT, BATCH), 256, 0, stream>>>(weight, style_ws, wmod);
    xcvt_kernel<<<(BATCH * CIN * NPIX) / (4 * 256), 256, 0, stream>>>(x, xbf);
    conv_wmma_kernel<<<dim3(NPIX / 256, COUT / 128, BATCH), 256, 0, stream>>>(
        xbf, wmod, smap, out);
}
